// FlashAttention_78486232367293
// MI455X (gfx1250) — matrix operand order repaired and checked by reading, not compiled, not run on hardware
//
#include <hip/hip_runtime.h>

typedef __attribute__((ext_vector_type(16))) _Float16 v16h;
typedef __attribute__((ext_vector_type(8)))  _Float16 v8h;
typedef __attribute__((ext_vector_type(2)))  _Float16 h2;
typedef __attribute__((ext_vector_type(8)))  float    v8f;

constexpr int B_ = 2, S_ = 2048, H_ = 16, D_ = 64;
constexpr int ROW  = H_ * D_;      // 1024 floats between consecutive seq positions
constexpr int KBLK = 64;           // keys per block iteration (4 x 16-col score tiles)
constexpr int WAVES = 8;
constexpr int QT = 16;             // query rows per wave
constexpr int BQ = WAVES * QT;     // 128 query rows per workgroup
// fold softmax scale and log2(e) into Q once: scores come out in log2 domain,
// so the hot loop uses raw v_exp_f32 (hardware exp2) with no per-element multiplies.
constexpr float SCALE_LOG2E = 0.125f * 1.44269504088896340736f;
constexpr float NEGBIG = -3.0e38f; // finite "-inf": exp2 underflows to 0, no NaN paths

__device__ __forceinline__ float exp2fast(float x) {
  return __builtin_amdgcn_exp2f(x);   // v_exp_f32
}

__device__ __forceinline__ v8f wmma16x16x32(v16h a, v16h b, v8f c) {
  return __builtin_amdgcn_wmma_f32_16x16x32_f16(false, a, false, b, (short)0, c,
                                                false, false);
}

// packed-f16 16-lane max butterfly: two rows per shuffle; f16 rounding of the
// stabilizer max is harmless (only needs >= max - eps).
__device__ __forceinline__ h2 rmax16_pk(h2 v) {
#pragma unroll
  for (int m = 1; m < 16; m <<= 1) {
    int yi = __shfl_xor(__builtin_bit_cast(int, v), m, 32);
    h2 y = __builtin_bit_cast(h2, yi);
    v[0] = (v[0] >= y[0]) ? v[0] : y[0];
    v[1] = (v[1] >= y[1]) ? v[1] : y[1];
  }
  return v;
}

__global__ __launch_bounds__(256) void
fa_fwd_78486232367293(const float* __restrict__ Q, const float* __restrict__ K,
                      const float* __restrict__ V,
                      const unsigned char* __restrict__ mask,
                      float* __restrict__ O) {
  __shared__ __align__(32) _Float16 sK[KBLK][D_];        //  8 KB key-row-major f16
  __shared__ __align__(32) _Float16 sVt[D_][KBLK];       //  8 KB transposed [d][key], masked rows zeroed
  __shared__ __align__(32) _Float16 sP[WAVES][QT][KBLK]; // 16 KB per-wave P scratch
  __shared__ __align__(32) _Float16 sM[KBLK];            // 0/1 key-mask fragment source

  const int tid  = threadIdx.x;
  const int lane = tid & 31;
  const int wv   = tid >> 5;
  const int ln16 = lane & 15;
  const int hh   = lane >> 4;   // half-wave select (A/B K-range, D row offset)

  const int bh = blockIdx.y;
  const int b = bh / H_, h = bh % H_;
  const int qbase = blockIdx.x * BQ;

  const size_t head = (size_t)b * S_ * ROW + (size_t)h * D_;
  const float* qh = Q + head;
  const float* kh = K + head;
  const float* vh = V + head;
  float*       oh = O + head;
  const unsigned char* mk = mask + (size_t)b * S_;

  // ---- Q A-fragments (16x32 f16 each, two K-steps over d=0..63), built once,
  //      pre-scaled by D^-0.5 * log2(e) ----
  v16h aq[2];
  {
    const float* qp = qh + (size_t)(qbase + wv * QT + ln16) * ROW;
#pragma unroll
    for (int ks = 0; ks < 2; ++ks) {
      const int d0 = ks * 32 + hh * 16;
      const int d1 = ks * 32 + hh * 16 + 8;
#pragma unroll
      for (int i = 0; i < 8; ++i) {
        aq[ks][i]     = (_Float16)(qp[d0 + i] * SCALE_LOG2E);
        aq[ks][8 + i] = (_Float16)(qp[d1 + i] * SCALE_LOG2E);
      }
    }
  }

  const v8f vzero = {0.f, 0.f, 0.f, 0.f, 0.f, 0.f, 0.f, 0.f};
  v8f acc[4];   // O accumulators, D-layout: lane -> d col (ln16), VGPR r -> row hh*8+r
#pragma unroll
  for (int t = 0; t < 4; ++t) acc[t] = vzero;
  v8f accL = vzero;  // denominator accumulator (row sums of masked P via WMMA)
  float mrow[8];
#pragma unroll
  for (int r = 0; r < 8; ++r) mrow[r] = NEGBIG;

  const int krow = tid >> 2;          // staging: key row 0..63
  const int dseg = (tid & 3) * 16;    // staging: 16-element d segment

  for (int kb = 0; kb < S_; kb += KBLK) {
    __syncthreads();  // protect sK/sVt/sM reuse across iterations
    // ---- cooperative stage: K row-major f16; V transposed f16 with masked
    //      key rows zeroed; 0/1 mask vector for the denominator fragment ----
    {
      const float* kp = kh + (size_t)(kb + krow) * ROW + dseg;
      const float* vp = vh + (size_t)(kb + krow) * ROW + dseg;
      const float vm = mk[kb + krow] ? 0.0f : 1.0f;   // zero masked V rows
      v8h k0, k1;
#pragma unroll
      for (int i = 0; i < 8; ++i) { k0[i] = (_Float16)kp[i]; k1[i] = (_Float16)kp[8 + i]; }
      *(v8h*)&sK[krow][dseg]     = k0;
      *(v8h*)&sK[krow][dseg + 8] = k1;
#pragma unroll
      for (int i = 0; i < 16; ++i) sVt[dseg + i][krow] = (_Float16)(vp[i] * vm);
      if (tid < KBLK) sM[tid] = mk[kb + tid] ? (_Float16)0.0f : (_Float16)1.0f;
      if (kb + KBLK < S_) {  // -> global_prefetch_b8 for next key block
        __builtin_prefetch(kp + (size_t)KBLK * ROW, 0, 1);
        __builtin_prefetch(vp + (size_t)KBLK * ROW, 0, 1);
      }
    }
    __syncthreads();

    // ---- scores S = (Q*log2e/sqrtD) * K^T : four 16x16 tiles over 64 keys ----
    v8f s[4];
#pragma unroll
    for (int t = 0; t < 4; ++t) s[t] = vzero;
#pragma unroll
    for (int ks = 0; ks < 2; ++ks) {
#pragma unroll
      for (int t = 0; t < 4; ++t) {
        v16h bk = *(const v16h*)&sK[t * 16 + ln16][ks * 32 + hh * 16];
        s[t] = wmma16x16x32(aq[ks], bk, s[t]);
      }
    }

    // ---- online softmax (log2 domain); masked keys need no score fixup:
    //      they are excluded via zeroed V rows + the sM denominator fragment ----
    float mx[8];
#pragma unroll
    for (int j = 0; j < 4; ++j) {
      h2 v;
      v[0] = (_Float16)fmaxf(fmaxf(s[0][2 * j],     s[1][2 * j]),
                             fmaxf(s[2][2 * j],     s[3][2 * j]));
      v[1] = (_Float16)fmaxf(fmaxf(s[0][2 * j + 1], s[1][2 * j + 1]),
                             fmaxf(s[2][2 * j + 1], s[3][2 * j + 1]));
      v = rmax16_pk(v);
      mx[2 * j]     = (float)v[0];
      mx[2 * j + 1] = (float)v[1];
    }
    const int rb = hh * 8;
#pragma unroll
    for (int r = 0; r < 8; ++r) {
      const float mn   = fmaxf(mrow[r], mx[r]);
      const float corr = exp2fast(mrow[r] - mn);
      mrow[r] = mn;
#pragma unroll
      for (int t = 0; t < 4; ++t) {
        const float p = exp2fast(s[t][r] - mn);
        sP[wv][rb + r][t * 16 + ln16] = (_Float16)p;  // P in f16, D-layout -> LDS
        acc[t][r] *= corr;
      }
      accL[r] *= corr;
    }
    asm volatile("s_wait_dscnt 0" ::: "memory");  // same-wave DS RAW fence

    // ---- reload P as A-fragments (16x64 -> two 16x32 K-steps) ----
    v16h pa[2];
#pragma unroll
    for (int ks = 0; ks < 2; ++ks) {
      const int c0 = ks * 32 + hh * 16;
      v8h lo = *(const v8h*)&sP[wv][ln16][c0];
      v8h hi = *(const v8h*)&sP[wv][ln16][c0 + 8];
#pragma unroll
      for (int i = 0; i < 8; ++i) { pa[ks][i] = lo[i]; pa[ks][8 + i] = hi[i]; }
    }

    // ---- O += P*V (masked V rows are zero) and l += P*mask01 via WMMA ----
#pragma unroll
    for (int ks = 0; ks < 2; ++ks) {
#pragma unroll
      for (int t = 0; t < 4; ++t) {
        v16h bv = *(const v16h*)&sVt[t * 16 + ln16][ks * 32 + hh * 16];
        acc[t] = wmma16x16x32(pa[ks], bv, acc[t]);
      }
      v16h mo = *(const v16h*)&sM[ks * 32 + hh * 16];
      accL = wmma16x16x32(pa[ks], mo, accL);
    }
  }

  // ---- normalize by row sums and store f32 output ----
#pragma unroll
  for (int r = 0; r < 8; ++r) {
    const float inv = 1.0f / accL[r];  // lane-uniform row sum (mask-WMMA columns)
    float* op = oh + (size_t)(qbase + wv * QT + hh * 8 + r) * ROW;
#pragma unroll
    for (int t = 0; t < 4; ++t) op[t * 16 + ln16] = acc[t][r] * inv;
  }
}

extern "C" void kernel_launch(void* const* d_in, const int* in_sizes, int n_in,
                              void* d_out, int out_size, void* d_ws, size_t ws_size,
                              hipStream_t stream) {
  (void)in_sizes; (void)n_in; (void)out_size; (void)d_ws; (void)ws_size;
  const float* q = (const float*)d_in[0];
  const float* k = (const float*)d_in[1];
  const float* v = (const float*)d_in[2];
  const unsigned char* mask = (const unsigned char*)d_in[3];  // bool -> 1 byte
  float* out = (float*)d_out;
  dim3 grid(S_ / BQ, B_ * H_);   // 16 x 32
  dim3 block(WAVES * 32);        // 8 waves (wave32)
  fa_fwd_78486232367293<<<grid, block, 0, stream>>>(q, k, v, mask, out);
}
